// MoLModel_20899310862740
// MI455X (gfx1250) — matrix operand order repaired and checked by reading, not compiled, not run on hardware
//
#include <hip/hip_runtime.h>
#include <hip/hip_bf16.h>

// ---------------------------------------------------------------------------
// MoL layer: out = x@W^T + b + softmax_e(q_e.k_e/sqrt(DK)) * (x@A_e^T)@Bm_e^T
// bf16 WMMA for all GEMMs; LoRA combine folded into the main GEMM as 128
// extra K-columns ([x|wh] @ [W|Bcat]^T, K=4224). Double-buffered LDS tiles
// fed by GLOBAL_LOAD_ASYNC_TO_LDS_B128 (ASYNCcnt-tracked DMA), overlapping
// tile DMA with the v_wmma_f32_16x16x32_bf16 block.
// ---------------------------------------------------------------------------

#define TOK   8192
#define DIN   4096
#define DOUT  4096
#define NE    8
#define DK    32
#define RR    16
#define KEXT  (NE * RR)                  // 128 extra K columns
#define KCAT  (DIN + KEXT)               // 4224
#define NROUTE (2 * NE * DK + NE * RR)   // 640

typedef __attribute__((ext_vector_type(16))) __bf16 v16bf;
typedef __attribute__((ext_vector_type(8)))  float  v8f;
typedef int v4i __attribute__((vector_size(16)));   // matches builtin param type

union FragBF { v16bf v; uint4 u[2]; };

// --- CDNA5 async global->LDS copy path (fallback: sync uint4 copy) ---------
#if __has_builtin(__builtin_amdgcn_global_load_async_to_lds_b128) && \
    __has_builtin(__builtin_amdgcn_s_wait_asynccnt)
#define USE_ASYNC_LDS 1
#endif

__device__ __forceinline__ void copy_b128_to_lds(void* lds_dst, const void* gsrc) {
#ifdef USE_ASYNC_LDS
    __builtin_amdgcn_global_load_async_to_lds_b128(
        (__attribute__((address_space(1))) v4i*)(v4i*)gsrc,
        (__attribute__((address_space(3))) v4i*)(v4i*)lds_dst,
        0 /*imm offset*/, 0 /*cpol*/);
#else
    *(uint4*)lds_dst = *(const uint4*)gsrc;
#endif
}

__device__ __forceinline__ void wait_async_tiles_prev() {
#ifdef USE_ASYNC_LDS
    __builtin_amdgcn_s_wait_asynccnt(5);   // 5 ops of the NEXT tile outstanding
#endif
}
__device__ __forceinline__ void wait_async_tiles_all() {
#ifdef USE_ASYNC_LDS
    __builtin_amdgcn_s_wait_asynccnt(0);
#endif
}

// ---------------------------------------------------------------------------
// Pack [W | Bcat] -> bf16 Wcat[DOUT][KCAT]
// ---------------------------------------------------------------------------
__global__ void prep_wcat_kernel(const float* __restrict__ W,
                                 const float* __restrict__ Bm,
                                 __bf16* __restrict__ Wcat) {
    const int o = blockIdx.x;
    for (int c = threadIdx.x; c < KCAT; c += blockDim.x) {
        float v;
        if (c < DIN) {
            v = W[(size_t)o * DIN + c];
        } else {
            const int j = c - DIN;                 // e*16 + r
            const int e = j >> 4, r = j & 15;
            v = Bm[((size_t)e * DOUT + o) * RR + r];
        }
        Wcat[(size_t)o * KCAT + c] = (__bf16)v;
    }
}

// ---------------------------------------------------------------------------
// Pack [Wq ; Wk ; A] -> bf16 Wroute[NROUTE][DIN]
// ---------------------------------------------------------------------------
__global__ void prep_wroute_kernel(const float* __restrict__ Wq,
                                   const float* __restrict__ Wk,
                                   const float* __restrict__ A,
                                   __bf16* __restrict__ Wroute) {
    const int n = blockIdx.x;
    const float* src;
    if (n < NE * DK)            src = Wq + (size_t)n * DIN;
    else if (n < 2 * NE * DK)   src = Wk + (size_t)(n - NE * DK) * DIN;
    else                        src = A  + (size_t)(n - 2 * NE * DK) * DIN;
    __bf16* dst = Wroute + (size_t)n * DIN;
    for (int c = threadIdx.x; c < DIN; c += blockDim.x)
        dst[c] = (__bf16)src[c];
}

// ---------------------------------------------------------------------------
// x fp32 -> bf16 into xcat[:, 0:DIN] (row stride KCAT). 4 elements / thread.
// ---------------------------------------------------------------------------
__global__ void cvt_x_kernel(const float* __restrict__ x,
                             __bf16* __restrict__ xcat) {
    const size_t idx = (size_t)blockIdx.x * blockDim.x + threadIdx.x;
    if (idx >= (size_t)TOK * (DIN / 4)) return;
    const int row  = (int)(idx >> 10);
    const int col4 = (int)(idx & 1023) * 4;
    const float4 v = *(const float4*)(x + (size_t)row * DIN + col4);
    union { __bf16 h[4]; uint2 u; } p;
    p.h[0] = (__bf16)v.x; p.h[1] = (__bf16)v.y;
    p.h[2] = (__bf16)v.z; p.h[3] = (__bf16)v.w;
    *(uint2*)(xcat + (size_t)row * KCAT + col4) = p.u;
}

// ---------------------------------------------------------------------------
// bf16 WMMA GEMM:  C[M,N] = A[M,K](lda) @ B[N,K](ldb)^T  (+ bias[N])
// 256 threads = 8 waves (wave32); workgroup tile 64x256; wave tile 32x64
// (2x4 WMMA 16x16 tiles). Double-buffered LDS, async tile DMA.
// K multiple of 32, M multiple of 64; N tail: clamp loads, predicate stores.
// ---------------------------------------------------------------------------
__global__ __launch_bounds__(256)
void gemm_bf16_wmma_kernel(const __bf16* __restrict__ A, int lda,
                           const __bf16* __restrict__ B, int ldb,
                           float* __restrict__ C, int ldc,
                           const float* __restrict__ bias,
                           int M, int N, int K) {
    __shared__ __align__(16) __bf16 sA[2][64 * 32];    // 2 x 4 KB
    __shared__ __align__(16) __bf16 sB[2][256 * 32];   // 2 x 16 KB

    const int lane = threadIdx.x & 31;
    const int wave = threadIdx.x >> 5;
    const int wm = wave & 1;        // 2 waves along M
    const int wn = wave >> 1;       // 4 waves along N
    const int blockM = blockIdx.y * 64;
    const int blockN = blockIdx.x * 256;

    // per-thread tile-copy coordinates (128-bit lanes)
    const int lr = threadIdx.x >> 2;            // 0..63
    const int lc = (threadIdx.x & 3) * 8;       // 0,8,16,24
    const __bf16* gA = A + (size_t)(blockM + lr) * lda + lc;
    const __bf16* gB0;
    const __bf16* gB1;
    const __bf16* gB2;
    const __bf16* gB3;
    {
        int n0 = blockN + lr;        if (n0 > N - 1) n0 = N - 1;
        int n1 = blockN + lr + 64;   if (n1 > N - 1) n1 = N - 1;
        int n2 = blockN + lr + 128;  if (n2 > N - 1) n2 = N - 1;
        int n3 = blockN + lr + 192;  if (n3 > N - 1) n3 = N - 1;
        gB0 = B + (size_t)n0 * ldb + lc;
        gB1 = B + (size_t)n1 * ldb + lc;
        gB2 = B + (size_t)n2 * ldb + lc;
        gB3 = B + (size_t)n3 * ldb + lc;
    }

    v8f acc[2][4];
    #pragma unroll
    for (int mi = 0; mi < 2; ++mi)
        #pragma unroll
        for (int ni = 0; ni < 4; ++ni)
            acc[mi][ni] = (v8f){0.f, 0.f, 0.f, 0.f, 0.f, 0.f, 0.f, 0.f};

    // per-lane fragment addressing (documented bf16 A/B VGPR layouts)
    const int mrow  = lane & 15;
    const int kOffA = (lane >> 4) * 8;    // A: lanes 16-31 hold K=8..15 / 24..31
    const int kOffB = (lane >> 4) * 8;    // B: lanes 16-31 hold K=8..15 / 24..31

    // issue tile copy for K-offset k0 into buffer `buf` (5 x b128 per thread)
    auto issue_tile = [&](int buf, int k0) {
        copy_b128_to_lds(&sA[buf][lr * 32 + lc],         gA  + k0);
        copy_b128_to_lds(&sB[buf][(lr      ) * 32 + lc], gB0 + k0);
        copy_b128_to_lds(&sB[buf][(lr +  64) * 32 + lc], gB1 + k0);
        copy_b128_to_lds(&sB[buf][(lr + 128) * 32 + lc], gB2 + k0);
        copy_b128_to_lds(&sB[buf][(lr + 192) * 32 + lc], gB3 + k0);
    };

    const int nt = K >> 5;          // number of K=32 tiles
    issue_tile(0, 0);               // prologue

    for (int t = 0; t < nt; ++t) {
        const int cur = t & 1;
        if (t + 1 < nt) {
            issue_tile(cur ^ 1, (t + 1) << 5);   // prefetch next tile
            wait_async_tiles_prev();             // tile t done (in-order), 5 pending
        } else {
            wait_async_tiles_all();
        }
        __syncthreads();                         // all waves' tile t in LDS

        // --- build fragments from LDS ---
        FragBF a[2], b[4];
        #pragma unroll
        for (int mi = 0; mi < 2; ++mi) {
            const __bf16* p = &sA[cur][(wm * 32 + mi * 16 + mrow) * 32];
            a[mi].u[0] = *(const uint4*)(p + kOffA);        // K  0..7 / 8..15
            a[mi].u[1] = *(const uint4*)(p + kOffA + 16);   // K 16..23 / 24..31
        }
        #pragma unroll
        for (int ni = 0; ni < 4; ++ni) {
            const __bf16* p =
                &sB[cur][(wn * 64 + ni * 16 + (lane & 15)) * 32 + kOffB];
            b[ni].u[0] = *(const uint4*)(p);                // K +0..7
            b[ni].u[1] = *(const uint4*)(p + 16);           // K +16..23
        }

        // --- 8x v_wmma_f32_16x16x32_bf16 per K step ---
        #pragma unroll
        for (int mi = 0; mi < 2; ++mi)
            #pragma unroll
            for (int ni = 0; ni < 4; ++ni)
                acc[mi][ni] = __builtin_amdgcn_wmma_f32_16x16x32_bf16(
                    false, a[mi].v, false, b[ni].v,
                    (short)0, acc[mi][ni], false, false);

        __syncthreads();            // everyone done reading buffer `cur`
    }

    // --- epilogue: C layout = lanes 0-15 rows j, lanes 16-31 rows j+8 ---
    const int mbase = blockM + wm * 32;
    const int nbase = blockN + wn * 64;
    #pragma unroll
    for (int mi = 0; mi < 2; ++mi) {
        #pragma unroll
        for (int ni = 0; ni < 4; ++ni) {
            const int col = nbase + ni * 16 + (lane & 15);
            if (col < N) {
                const float bv = bias ? bias[col] : 0.0f;
                #pragma unroll
                for (int j = 0; j < 8; ++j) {
                    const int row = mbase + mi * 16 + (lane >> 4) * 8 + j;
                    C[(size_t)row * ldc + col] = acc[mi][ni][j] + bv;
                }
            }
        }
    }
}

// ---------------------------------------------------------------------------
// Per-token routing: scores_e = q_e.k_e / sqrt(DK); softmax over E;
// wh[e*16+r] = w_e * h[e*16+r] * SCALING  -> bf16 into xcat[:, DIN:KCAT]
// T row layout: [0:256)=q, [256:512)=k, [512:640)=h.
// ---------------------------------------------------------------------------
__global__ void route_softmax_kernel(const float* __restrict__ T,
                                     __bf16* __restrict__ xcat) {
    const int tok = blockIdx.x * blockDim.x + threadIdx.x;
    if (tok >= TOK) return;
    const float* t = T + (size_t)tok * NROUTE;
    const float inv_sqrt_dk = 0.17677669529663687f;   // 1/sqrt(32)

    float s[NE];
    float smax = -3.0e38f;
    #pragma unroll
    for (int e = 0; e < NE; ++e) {
        float accq = 0.f;
        #pragma unroll
        for (int d = 0; d < DK; ++d)
            accq += t[e * DK + d] * t[NE * DK + e * DK + d];
        s[e] = accq * inv_sqrt_dk;
        smax = fmaxf(smax, s[e]);
    }
    float ssum = 0.f;
    #pragma unroll
    for (int e = 0; e < NE; ++e) { s[e] = __expf(s[e] - smax); ssum += s[e]; }
    const float inv = 1.0f / ssum;

    const float scaling = 1.0f;   // lora_alpha / r = 16/16
    __bf16* dst = xcat + (size_t)tok * KCAT + DIN;
    #pragma unroll
    for (int e = 0; e < NE; ++e) {
        const float we = s[e] * inv * scaling;
        #pragma unroll
        for (int r = 0; r < RR; ++r)
            dst[e * RR + r] = (__bf16)(we * t[2 * NE * DK + e * RR + r]);
    }
}

// ---------------------------------------------------------------------------
extern "C" void kernel_launch(void* const* d_in, const int* in_sizes, int n_in,
                              void* d_out, int out_size, void* d_ws, size_t ws_size,
                              hipStream_t stream) {
    const float* x  = (const float*)d_in[0];   // [4,2048,4096]
    const float* W  = (const float*)d_in[1];   // [4096,4096]
    const float* b  = (const float*)d_in[2];   // [4096]
    const float* Wq = (const float*)d_in[3];   // [256,4096]
    const float* Wk = (const float*)d_in[4];   // [256,4096]
    const float* A  = (const float*)d_in[5];   // [8,16,4096]
    const float* Bm = (const float*)d_in[6];   // [8,4096,16]
    float* out = (float*)d_out;                // [4,2048,4096]

    // workspace carving
    char* ws = (char*)d_ws;
    __bf16* xcat   = (__bf16*)ws;                                   // 8192*4224*2
    size_t off = (size_t)TOK * KCAT * 2;
    __bf16* Wcat   = (__bf16*)(ws + off);                           // 4096*4224*2
    off += (size_t)DOUT * KCAT * 2;
    __bf16* Wroute = (__bf16*)(ws + off);                           // 640*4096*2
    off += (size_t)NROUTE * DIN * 2;
    float*  T      = (float*)(ws + off);                            // 8192*640*4

    // 1. pack weights to bf16
    prep_wcat_kernel<<<DOUT, 256, 0, stream>>>(W, Bm, Wcat);
    prep_wroute_kernel<<<NROUTE, 256, 0, stream>>>(Wq, Wk, A, Wroute);

    // 2. x -> bf16 (cols 0..DIN of xcat)
    {
        const size_t total = (size_t)TOK * (DIN / 4);
        cvt_x_kernel<<<(int)((total + 255) / 256), 256, 0, stream>>>(x, xcat);
    }

    // 3. routing GEMM: T[8192,640] = xcat(:,:4096) @ Wroute^T
    {
        dim3 grid((NROUTE + 255) / 256, TOK / 64);
        gemm_bf16_wmma_kernel<<<grid, 256, 0, stream>>>(
            xcat, KCAT, Wroute, DIN, T, NROUTE, nullptr, TOK, NROUTE, DIN);
    }

    // 4. softmax + weighted h -> xcat[:, DIN:]
    route_softmax_kernel<<<TOK / 256, 256, 0, stream>>>(T, xcat);

    // 5. main GEMM (LoRA folded into K): out = xcat @ Wcat^T + b
    {
        dim3 grid(DOUT / 256, TOK / 64);
        gemm_bf16_wmma_kernel<<<grid, 256, 0, stream>>>(
            xcat, KCAT, Wcat, KCAT, out, DOUT, b, TOK, DOUT, KCAT);
    }
}
